// CrossAttention_80590766342194
// MI455X (gfx1250) — matrix operand order repaired and checked by reading, not compiled, not run on hardware
//
#include <hip/hip_runtime.h>

typedef __attribute__((ext_vector_type(16))) __bf16 v16bf;
typedef __attribute__((ext_vector_type(8)))  float  v8f;
typedef __attribute__((ext_vector_type(4)))  int    v4i;

#define NQ   4096
#define NYK  4096
#define NH   8
#define HD   64
#define CDEC 512
#define CENC 768
#define MROWS 8192          // B * N  (= B * NY)
#define ATT_SCALE 0.125f    // 1/sqrt(64)

// ---------- CDNA5 async global->LDS staging (guarded) ----------

#if defined(__has_builtin)
#if __has_builtin(__builtin_amdgcn_global_load_async_to_lds_b128)
#define HAVE_ASYNC_LDS 1
#endif
#endif

__device__ __forceinline__ void cp16(unsigned short* lds_dst, const unsigned short* g_src) {
#ifdef HAVE_ASYNC_LDS
    __builtin_amdgcn_global_load_async_to_lds_b128(
        (__attribute__((address_space(1))) v4i*)g_src,
        (__attribute__((address_space(3))) v4i*)lds_dst, 0, 0);
#else
    *(uint4*)lds_dst = *(const uint4*)g_src;
#endif
}

__device__ __forceinline__ void staging_fence() {
#ifdef HAVE_ASYNC_LDS
    asm volatile("s_wait_asynccnt 0x0" ::: "memory");
#endif
}

// ---------- helpers ----------

__device__ __forceinline__ unsigned short f2bf(float f) {
    union { float f; unsigned u; } v; v.f = f;
    unsigned r = v.u + 0x7FFFu + ((v.u >> 16) & 1u);
    return (unsigned short)(r >> 16);
}

__device__ __forceinline__ v8f wmma_bf16(v16bf a, v16bf b, v8f c) {
    return __builtin_amdgcn_wmma_f32_16x16x32_bf16(
        false, a, false, b, (short)0, c, false, false);
}

// A fragment (16x32 bf16) from row-major LDS tile (stride in u16 units).
__device__ __forceinline__ v16bf load_a_frag(const unsigned short* base, int stride, int lane) {
    int row = lane & 15;
    int kb  = (lane & 16) ? 8 : 0;
    const unsigned short* p = base + row * stride + kb;
    union { uint4 q[2]; v16bf v; } u;
    u.q[0] = *(const uint4*)(p);        // K = kb .. kb+7
    u.q[1] = *(const uint4*)(p + 16);   // K = kb+16 .. kb+23
    return u.v;
}

// B fragment (32x16 bf16) from an n-major tile: tile[n][k].
__device__ __forceinline__ v16bf load_b_frag(const unsigned short* base, int stride, int lane) {
    int n  = lane & 15;
    int kb = (lane & 16) ? 8 : 0;
    const unsigned short* p = base + n * stride + kb;
    union { uint4 q[2]; v16bf v; } u;
    u.q[0] = *(const uint4*)(p);
    u.q[1] = *(const uint4*)(p + 16);
    return u.v;
}

// ---------- fp32 -> bf16 convert (x4 vectorized) ----------

__global__ __launch_bounds__(256) void cvt_f32_bf16_x4(const float4* __restrict__ src,
                                                       ushort4* __restrict__ dst, int n4) {
    int i = blockIdx.x * blockDim.x + threadIdx.x;
    if (i < n4) {
        float4 f = src[i];
        ushort4 r;
        r.x = f2bf(f.x); r.y = f2bf(f.y); r.z = f2bf(f.z); r.w = f2bf(f.w);
        dst[i] = r;
    }
}

// ---------- WMMA GEMM: C[M,Nn] = A[M,K](bf16) @ W[Nn,K](bf16)^T ----------
// 128x64 tile per block (4 waves); each wave computes 32x64 (2 A-frags reuse B-frags).
// mode 0: scatter bf16 -> Q layout [B,H,N,64]
// mode 1: scatter bf16 -> K layout [B,H,NY,64] and V TRANSPOSED layout [B,H,64,NY]
// mode 2: fp32 out[m*512+c] = acc + bias[c]

__global__ __launch_bounds__(128) void gemm_bf16_kernel(
    const unsigned short* __restrict__ A,
    const unsigned short* __restrict__ W,
    int K, int mode,
    unsigned short* __restrict__ outQ,
    unsigned short* __restrict__ outK,
    unsigned short* __restrict__ outV,
    const float* __restrict__ bias,
    float* __restrict__ outF)
{
    __shared__ unsigned short Al[128 * 40];  // 128 rows x 32 k, stride 40
    __shared__ unsigned short Bl[64 * 40];   // 64 n-rows x 32 k, stride 40

    const int tid  = threadIdx.x;
    const int lane = tid & 31;
    const int wave = tid >> 5;
    const int m0 = blockIdx.x * 128;
    const int n0 = blockIdx.y * 64;

    v8f acc[2][4] = { { {}, {}, {}, {} }, { {}, {}, {}, {} } };

    for (int kc = 0; kc < K; kc += 32) {
        {   // A tile: one full 32-half row per thread
            const unsigned short* ga = A + (size_t)(m0 + tid) * K + kc;
            unsigned short* sa = &Al[tid * 40];
            cp16(sa, ga); cp16(sa + 8, ga + 8); cp16(sa + 16, ga + 16); cp16(sa + 24, ga + 24);
            // B tile: half row per thread
            int rb = tid >> 1, cb = (tid & 1) * 16;
            const unsigned short* gb = W + (size_t)(n0 + rb) * K + kc + cb;
            unsigned short* sb = &Bl[rb * 40 + cb];
            cp16(sb, gb); cp16(sb + 8, gb + 8);
        }
        staging_fence();
        __syncthreads();

        v16bf af0 = load_a_frag(Al + (wave * 32) * 40,      40, lane);
        v16bf af1 = load_a_frag(Al + (wave * 32 + 16) * 40, 40, lane);
#pragma unroll
        for (int jj = 0; jj < 4; ++jj) {
            v16bf bf = load_b_frag(Bl + jj * 16 * 40, 40, lane);
            acc[0][jj] = wmma_bf16(af0, bf, acc[0][jj]);
            acc[1][jj] = wmma_bf16(af1, bf, acc[1][jj]);
        }
        __syncthreads();
    }

    const int hi8 = (lane & 16) ? 8 : 0;
    const int cn  = lane & 15;
#pragma unroll
    for (int g = 0; g < 2; ++g) {
#pragma unroll
        for (int jj = 0; jj < 4; ++jj) {
#pragma unroll
            for (int i = 0; i < 8; ++i) {
                int row = m0 + wave * 32 + g * 16 + i + hi8;  // global M row
                int col = n0 + jj * 16 + cn;                  // global N col
                float v = acc[g][jj][i];
                if (mode == 0) {
                    int b = row >> 12, n = row & 4095;
                    int h = col >> 6,  d = col & 63;
                    outQ[((size_t)((b * NH + h) * NQ + n)) * HD + d] = f2bf(v);
                } else if (mode == 1) {
                    int b = row >> 12, n = row & 4095;
                    if (col < CDEC) {
                        int h = col >> 6, d = col & 63;
                        outK[((size_t)((b * NH + h) * NYK + n)) * HD + d] = f2bf(v);
                    } else {
                        int c2 = col - CDEC;
                        int h = c2 >> 6, d = c2 & 63;
                        // V stored transposed per head: [B,H,64(d),NY]
                        outV[((size_t)((b * NH + h) * HD + d)) * NYK + n] = f2bf(v);
                    }
                } else {
                    outF[(size_t)row * CDEC + col] = v + bias[col];
                }
            }
        }
    }
}

// ---------- flash attention ----------
// grid (N/64, B*H), 128 threads (4 waves); each wave owns 16 Q rows.
// 64-key tiles: 16 WMMAs per wave per tile, 3 barriers per tile.

__global__ __launch_bounds__(128) void attn_kernel(
    const unsigned short* __restrict__ Qws,   // [B*H, N, 64] bf16
    const unsigned short* __restrict__ Kws,   // [B*H, NY, 64] bf16
    const unsigned short* __restrict__ VTws,  // [B*H, 64, NY] bf16 (pre-transposed)
    const unsigned char*  __restrict__ mask,  // [B, NY] bool
    unsigned short*       __restrict__ AO)    // [B*N, 512] bf16
{
    __shared__ unsigned short Ql[64 * 72];    // 64 q-rows x 64 d
    __shared__ unsigned short Kl[64 * 72];    // 64 keys  x 64 d   (n-major for S)
    __shared__ unsigned short Vt[64 * 72];    // 64 d     x 64 keys (n-major for PV)
    __shared__ unsigned short Pl[4][16 * 72]; // per-wave P scratch 16x64

    const int bh = blockIdx.y;
    const int bq = bh >> 3;
    const int h  = bh & 7;
    const int m0 = blockIdx.x * 64;

    const unsigned short* Qb = Qws  + ((size_t)bh * NQ + m0) * HD;
    const unsigned short* Kb = Kws  + (size_t)bh * NYK * HD;
    const unsigned short* Vb = VTws + (size_t)bh * HD * NYK;
    const unsigned char*  mk = mask + (size_t)bq * NYK;

    const int tid  = threadIdx.x;
    const int lane = tid & 31;
    const int wave = tid >> 5;
    const int cn   = lane & 15;
    const int hi8  = (lane & 16) ? 8 : 0;

    // stage Q tile (64x64)
    {
        int r  = tid >> 1;
        int cb = (tid & 1) * 32;
        const unsigned short* g = Qb + (size_t)r * HD + cb;
        unsigned short* s = &Ql[r * 72 + cb];
        cp16(s, g); cp16(s + 8, g + 8); cp16(s + 16, g + 16); cp16(s + 24, g + 24);
    }
    staging_fence();
    __syncthreads();

    const v16bf qf0 = load_a_frag(Ql + wave * 16 * 72,      72, lane);  // d = 0..31
    const v16bf qf1 = load_a_frag(Ql + wave * 16 * 72 + 32, 72, lane);  // d = 32..63

    v8f o[4] = { {}, {}, {}, {} };
    float mrow[8], lrow[8];
#pragma unroll
    for (int i = 0; i < 8; ++i) { mrow[i] = -__builtin_inff(); lrow[i] = 0.0f; }

    for (int t0 = 0; t0 < NYK; t0 += 64) {
        // stage K tile [64 keys][64 d] and Vt tile [64 d][64 keys]
        {
            int r  = tid >> 1;
            int cb = (tid & 1) * 32;
            const unsigned short* gk = Kb + (size_t)(t0 + r) * HD + cb;
            unsigned short* sk = &Kl[r * 72 + cb];
            cp16(sk, gk); cp16(sk + 8, gk + 8); cp16(sk + 16, gk + 16); cp16(sk + 24, gk + 24);
            const unsigned short* gv = Vb + (size_t)r * NYK + t0 + cb;
            unsigned short* sv = &Vt[r * 72 + cb];
            cp16(sv, gv); cp16(sv + 8, gv + 8); cp16(sv + 16, gv + 16); cp16(sv + 24, gv + 24);
        }
        staging_fence();
        __syncthreads();

        // S = Q @ K^T : 4 col fragments x (d split 2x32)
        v8f s[4] = { {}, {}, {}, {} };
#pragma unroll
        for (int jj = 0; jj < 4; ++jj) {
            v16bf b0 = load_b_frag(Kl + jj * 16 * 72,      72, lane);
            s[jj] = wmma_bf16(qf0, b0, s[jj]);
            v16bf b1 = load_b_frag(Kl + jj * 16 * 72 + 32, 72, lane);
            s[jj] = wmma_bf16(qf1, b1, s[jj]);
        }

        bool msk[4];
#pragma unroll
        for (int jj = 0; jj < 4; ++jj) msk[jj] = mk[t0 + jj * 16 + cn] != 0;

        // online softmax (rows split lanes 0-15 / 16-31; butterfly stays in half)
#pragma unroll
        for (int i = 0; i < 8; ++i) {
            float a[4];
#pragma unroll
            for (int jj = 0; jj < 4; ++jj)
                a[jj] = msk[jj] ? -__builtin_inff() : s[jj][i] * ATT_SCALE;
            float mx = fmaxf(fmaxf(a[0], a[1]), fmaxf(a[2], a[3]));
            mx = fmaxf(mx, __shfl_xor(mx, 1, 32));
            mx = fmaxf(mx, __shfl_xor(mx, 2, 32));
            mx = fmaxf(mx, __shfl_xor(mx, 4, 32));
            mx = fmaxf(mx, __shfl_xor(mx, 8, 32));
            float mnew = fmaxf(mrow[i], mx);
            float mb   = (mnew == -__builtin_inff()) ? 0.0f : mnew;
            float alpha = __expf(fmaxf(mrow[i], -3.0e38f) - mb);
            float rs = 0.0f;
            float p[4];
#pragma unroll
            for (int jj = 0; jj < 4; ++jj) { p[jj] = __expf(a[jj] - mb); rs += p[jj]; }
            rs += __shfl_xor(rs, 1, 32);
            rs += __shfl_xor(rs, 2, 32);
            rs += __shfl_xor(rs, 4, 32);
            rs += __shfl_xor(rs, 8, 32);
            lrow[i] = lrow[i] * alpha + rs;
            mrow[i] = mnew;
            o[0][i] *= alpha; o[1][i] *= alpha; o[2][i] *= alpha; o[3][i] *= alpha;
#pragma unroll
            for (int jj = 0; jj < 4; ++jj)
                Pl[wave][(i + hi8) * 72 + jj * 16 + cn] = f2bf(p[jj]);
        }
        __syncthreads();

        // O += P @ V : A = P 16x64 (2 frags), B = Vt (n-major over d columns)
        v16bf pf0 = load_a_frag(Pl[wave],      72, lane);
        v16bf pf1 = load_a_frag(Pl[wave] + 32, 72, lane);
#pragma unroll
        for (int jj = 0; jj < 4; ++jj) {
            o[jj] = wmma_bf16(pf0, load_b_frag(Vt + jj * 16 * 72,      72, lane), o[jj]);
            o[jj] = wmma_bf16(pf1, load_b_frag(Vt + jj * 16 * 72 + 32, 72, lane), o[jj]);
        }
        __syncthreads();
    }

    // normalize and scatter to AO [B*N, 512] (col = h*64 + d)
    const size_t mbase = (size_t)bq * NQ + m0 + wave * 16;
#pragma unroll
    for (int i = 0; i < 8; ++i) {
        float rinv = (lrow[i] > 0.0f) ? (1.0f / lrow[i]) : 0.0f;
        size_t row = mbase + i + hi8;
#pragma unroll
        for (int jj = 0; jj < 4; ++jj) {
            AO[row * CDEC + h * HD + jj * 16 + cn] = f2bf(o[jj][i] * rinv);
        }
    }
}

// ---------- host ----------

extern "C" void kernel_launch(void* const* d_in, const int* in_sizes, int n_in,
                              void* d_out, int out_size, void* d_ws, size_t ws_size,
                              hipStream_t stream) {
    const float*         x    = (const float*)d_in[0];
    const float*         y    = (const float*)d_in[1];
    const unsigned char* mask = (const unsigned char*)d_in[2];
    const float*         Wq   = (const float*)d_in[3];
    const float*         Wkv  = (const float*)d_in[4];
    const float*         Wp   = (const float*)d_in[5];
    const float*         bp   = (const float*)d_in[6];
    float*               out  = (float*)d_out;

    unsigned short* ws = (unsigned short*)d_ws;
    size_t off = 0;
    unsigned short* xbf  = ws + off; off += (size_t)MROWS * CDEC;      // 8192x512
    unsigned short* ybf  = ws + off; off += (size_t)MROWS * CENC;      // 8192x768
    unsigned short* wqb  = ws + off; off += (size_t)CDEC * CDEC;       // 512x512
    unsigned short* wkvb = ws + off; off += (size_t)(2 * CDEC) * CENC; // 1024x768
    unsigned short* wpb  = ws + off; off += (size_t)CDEC * CDEC;       // 512x512
    unsigned short* Qws  = ws + off; off += (size_t)MROWS * CDEC;      // [B,H,N,64]
    unsigned short* Kws  = ws + off; off += (size_t)MROWS * CDEC;      // [B,H,NY,64]
    unsigned short* VTws = ws + off; off += (size_t)MROWS * CDEC;      // [B,H,64,NY]
    unsigned short* AO   = ws + off; off += (size_t)MROWS * CDEC;      // [B*N,512]

    auto cvt = [&](const float* s, unsigned short* d, int n) {
        int n4 = n >> 2;
        cvt_f32_bf16_x4<<<(n4 + 255) / 256, 256, 0, stream>>>(
            (const float4*)s, (ushort4*)d, n4);
    };
    cvt(x,   xbf,  MROWS * CDEC);
    cvt(y,   ybf,  MROWS * CENC);
    cvt(Wq,  wqb,  CDEC * CDEC);
    cvt(Wkv, wkvb, 2 * CDEC * CENC);
    cvt(Wp,  wpb,  CDEC * CDEC);

    // Q = x @ Wq^T  -> Qws  (M=8192, Nn=512, K=512)
    gemm_bf16_kernel<<<dim3(MROWS / 128, CDEC / 64), 128, 0, stream>>>(
        xbf, wqb, CDEC, 0, Qws, nullptr, nullptr, nullptr, nullptr);
    // KV = y @ Wkv^T -> Kws / VTws (M=8192, Nn=1024, K=768)
    gemm_bf16_kernel<<<dim3(MROWS / 128, (2 * CDEC) / 64), 128, 0, stream>>>(
        ybf, wkvb, CENC, 1, nullptr, Kws, VTws, nullptr, nullptr);
    // flash attention -> AO
    attn_kernel<<<dim3(NQ / 64, 2 * NH), 128, 0, stream>>>(Qws, Kws, VTws, mask, AO);
    // out = AO @ Wp^T + bp (fp32)
    gemm_bf16_kernel<<<dim3(MROWS / 128, CDEC / 64), 128, 0, stream>>>(
        AO, wpb, CDEC, 2, nullptr, nullptr, nullptr, bp, out);
}
